// FeedForwardExperts_72816875536876
// MI455X (gfx1250) — matrix operand order repaired and checked by reading, not compiled, not run on hardware
//
#include <hip/hip_runtime.h>
#include <math.h>

// ---------------------------------------------------------------------------
// FeedForwardExperts (MoE FFN): x = gelu(in @ wi + bi) @ wo + bo, per expert.
// G=8, E=8, C=512, H=1024, F=4096, all f32 in HBM.
// Two bf16-WMMA GEMM kernels; intermediate gecf kept as bf16 in scratch
// (4 MB per (g,e) unit, chunked to fit ws_size). Mainloops are software-
// pipelined with double-buffered LDS: global loads for tile t+1 issue before
// the WMMAs of tile t, one barrier per K-step.
// ---------------------------------------------------------------------------

typedef __attribute__((ext_vector_type(16))) __bf16 v16bf;
typedef __attribute__((ext_vector_type(8)))  __bf16 v8bf;
typedef __attribute__((ext_vector_type(4)))  __bf16 v4bf;
typedef __attribute__((ext_vector_type(8)))  float  v8f;

#define CC 512
#define HH 1024
#define FF 4096
#define EE 8

#define BM 128
#define BN 128
#define BK 32
#define A_STRIDE 40    // BK + 8 bf16 pad: conflict-free rows, 16B-aligned
#define B_STRIDE 136   // BN + 8 bf16 pad
#define A_BUF (BM * A_STRIDE)
#define B_BUF (BK * B_STRIDE)

__device__ __forceinline__ float gelu_exact(float x) {
    return 0.5f * x * (1.0f + erff(x * 0.70710678118654752440f));
}

__device__ __forceinline__ v16bf concat16(v8bf lo, v8bf hi) {
    return __builtin_shufflevector(lo, hi, 0, 1, 2, 3, 4, 5, 6, 7,
                                           8, 9, 10, 11, 12, 13, 14, 15);
}

// A fragment (16-bit A 16x32 per ISA): lane L holds row M=L&15;
// halves 0..7 -> K = g*8..g*8+7, halves 8..15 -> K = g*8+16..g*8+23, g=L>>4.
__device__ __forceinline__ v16bf load_a_frag(const __bf16* Al, int m, int g16) {
    const __bf16* p = Al + m * A_STRIDE + g16 * 8;
    return concat16(*(const v8bf*)p, *(const v8bf*)(p + 16));
}

// B fragment (16-bit B 32x16 per ISA): lane L holds column N=L&15;
// halves 0..7 -> K = g*8..g*8+7, halves 8..15 -> K = g*8+16..g*8+23, g=L>>4.
__device__ __forceinline__ v16bf load_b_frag(const __bf16* Bl, int lane, int n) {
    const __bf16* p = Bl + (lane >> 4) * 8 * B_STRIDE + n + (lane & 15);
    v16bf b;
    #pragma unroll
    for (int s = 0; s < 8; ++s) {
        b[s]     = p[s * B_STRIDE];
        b[s + 8] = p[(s + 16) * B_STRIDE];
    }
    return b;
}

__device__ __forceinline__ void cvt_store4(__bf16* dst, float4 f) {
    v4bf p;
    p[0] = (__bf16)f.x; p[1] = (__bf16)f.y;
    p[2] = (__bf16)f.z; p[3] = (__bf16)f.w;
    *(v4bf*)dst = p;
}

// ---------------------------------------------------------------------------
// Kernel 1: ws[z][c][f] = bf16( gelu( inp[u][c][:] @ wi[e][:][f] + bi[e][f] ) )
// M=C, N=F, K=H.
// ---------------------------------------------------------------------------
__global__ __launch_bounds__(256)
void ffn_gemm1_gelu(const float* __restrict__ inp,
                    const float* __restrict__ wi,
                    const float* __restrict__ bi,
                    __bf16* __restrict__ ws,
                    int u_base) {
    __shared__ __bf16 Al[2 * A_BUF];
    __shared__ __bf16 Bl[2 * B_BUF];

    const int u = u_base + (int)blockIdx.z;
    const int e = u & (EE - 1);
    const float* A = inp + (size_t)u * CC * HH;   // [C][H]
    const float* B = wi  + (size_t)e * HH * FF;   // [H][F]
    const int n0 = blockIdx.x * BN;
    const int m0 = blockIdx.y * BM;

    const int tid  = threadIdx.x;
    const int lane = tid & 31;
    const int wave = tid >> 5;
    const int wr   = wave >> 2;   // 0..1 over M
    const int wc   = wave & 3;    // 0..3 over N
    const int g16  = lane >> 4;
    const int l16  = lane & 15;

    const int ar = tid >> 3;        // A loader: 32 rows/pass
    const int ak = (tid & 7) * 4;   // 4 k-floats per thread
    const int bk = tid >> 3;        // B loader: row k
    const int bn = (tid & 7) * 4;   // 4 n-floats per thread, 4 passes

    v8f acc[4][2] = {};
    float4 aReg[4], bReg[4];

    auto load_tile = [&](int k0) {
        #pragma unroll
        for (int r = 0; r < 4; ++r)
            aReg[r] = *(const float4*)(A + (size_t)(m0 + ar + r * 32) * HH + (k0 + ak));
        #pragma unroll
        for (int r = 0; r < 4; ++r)
            bReg[r] = *(const float4*)(B + (size_t)(k0 + bk) * FF + (n0 + bn + r * 32));
    };
    auto store_tile = [&](int buf) {
        __bf16* Ab = Al + buf * A_BUF;
        __bf16* Bb = Bl + buf * B_BUF;
        #pragma unroll
        for (int r = 0; r < 4; ++r)
            cvt_store4(&Ab[(ar + r * 32) * A_STRIDE + ak], aReg[r]);
        #pragma unroll
        for (int r = 0; r < 4; ++r)
            cvt_store4(&Bb[bk * B_STRIDE + bn + r * 32], bReg[r]);
    };
    auto compute = [&](int buf) {
        const __bf16* Ab = Al + buf * A_BUF;
        const __bf16* Bb = Bl + buf * B_BUF;
        v16bf afr[4];
        #pragma unroll
        for (int i = 0; i < 4; ++i)
            afr[i] = load_a_frag(Ab, wr * 64 + i * 16 + l16, g16);
        v16bf bfr[2];
        #pragma unroll
        for (int j = 0; j < 2; ++j)
            bfr[j] = load_b_frag(Bb, lane, wc * 32 + j * 16);
        #pragma unroll
        for (int i = 0; i < 4; ++i)
            #pragma unroll
            for (int j = 0; j < 2; ++j)
                acc[i][j] = __builtin_amdgcn_wmma_f32_16x16x32_bf16(
                    false, afr[i], false, bfr[j], (short)0, acc[i][j], false, false);
    };

    // Software pipeline: prime buf 0, then 1 barrier per K-step.
    const int NK = HH / BK;
    load_tile(0);
    store_tile(0);
    __syncthreads();
    for (int t = 0; t < NK; ++t) {
        const int cur = t & 1;
        if (t + 1 < NK) load_tile((t + 1) * BK);   // overlap with compute below
        compute(cur);
        if (t + 1 < NK) {
            store_tile(1 - cur);                   // buf read last at t-1: safe
            __syncthreads();
        }
    }

    // Epilogue: bias + exact gelu -> bf16 scratch.
    #pragma unroll
    for (int j = 0; j < 2; ++j) {
        const int nloc = wc * 32 + j * 16 + l16;
        const float bias = bi[(size_t)e * FF + n0 + nloc];
        #pragma unroll
        for (int i = 0; i < 4; ++i) {
            #pragma unroll
            for (int v = 0; v < 8; ++v) {
                const int mloc = wr * 64 + i * 16 + g16 * 8 + v;
                const float x = acc[i][j][v] + bias;
                ws[((size_t)blockIdx.z * CC + (m0 + mloc)) * FF + (n0 + nloc)] =
                    (__bf16)gelu_exact(x);
            }
        }
    }
}

// ---------------------------------------------------------------------------
// Kernel 2: out[u][c][h] = ws[z][c][:] @ wo[e][:][h] + bo[e][h]
// M=C, N=H, K=F. A is already bf16.
// ---------------------------------------------------------------------------
__global__ __launch_bounds__(256)
void ffn_gemm2(const __bf16* __restrict__ ws,
               const float* __restrict__ wo,
               const float* __restrict__ bo,
               float* __restrict__ out,
               int u_base) {
    __shared__ __bf16 Al[2 * A_BUF];
    __shared__ __bf16 Bl[2 * B_BUF];

    const int u = u_base + (int)blockIdx.z;
    const int e = u & (EE - 1);
    const __bf16* A = ws + (size_t)blockIdx.z * CC * FF;  // [C][F] bf16
    const float*  B = wo + (size_t)e * FF * HH;           // [F][H]
    const int n0 = blockIdx.x * BN;
    const int m0 = blockIdx.y * BM;

    const int tid  = threadIdx.x;
    const int lane = tid & 31;
    const int wave = tid >> 5;
    const int wr   = wave >> 2;
    const int wc   = wave & 3;
    const int g16  = lane >> 4;
    const int l16  = lane & 15;

    const int ar2 = tid >> 1;          // 0..127: one LDS row per thread
    const int ac2 = (tid & 1) * 16;    // half of the 32-wide k slab
    const int bk  = tid >> 3;
    const int bn  = (tid & 7) * 4;

    v8f acc[4][2] = {};
    v8bf aReg[2];
    float4 bReg[4];

    auto load_tile = [&](int k0) {
        const __bf16* src = A + (size_t)(m0 + ar2) * FF + (k0 + ac2);
        aReg[0] = *(const v8bf*)src;
        aReg[1] = *(const v8bf*)(src + 8);
        #pragma unroll
        for (int r = 0; r < 4; ++r)
            bReg[r] = *(const float4*)(B + (size_t)(k0 + bk) * HH + (n0 + bn + r * 32));
    };
    auto store_tile = [&](int buf) {
        __bf16* Ab = Al + buf * A_BUF;
        __bf16* Bb = Bl + buf * B_BUF;
        *(v8bf*)&Ab[ar2 * A_STRIDE + ac2]     = aReg[0];
        *(v8bf*)&Ab[ar2 * A_STRIDE + ac2 + 8] = aReg[1];
        #pragma unroll
        for (int r = 0; r < 4; ++r)
            cvt_store4(&Bb[bk * B_STRIDE + bn + r * 32], bReg[r]);
    };
    auto compute = [&](int buf) {
        const __bf16* Ab = Al + buf * A_BUF;
        const __bf16* Bb = Bl + buf * B_BUF;
        v16bf afr[4];
        #pragma unroll
        for (int i = 0; i < 4; ++i)
            afr[i] = load_a_frag(Ab, wr * 64 + i * 16 + l16, g16);
        v16bf bfr[2];
        #pragma unroll
        for (int j = 0; j < 2; ++j)
            bfr[j] = load_b_frag(Bb, lane, wc * 32 + j * 16);
        #pragma unroll
        for (int i = 0; i < 4; ++i)
            #pragma unroll
            for (int j = 0; j < 2; ++j)
                acc[i][j] = __builtin_amdgcn_wmma_f32_16x16x32_bf16(
                    false, afr[i], false, bfr[j], (short)0, acc[i][j], false, false);
    };

    const int NK = FF / BK;
    load_tile(0);
    store_tile(0);
    __syncthreads();
    for (int t = 0; t < NK; ++t) {
        const int cur = t & 1;
        if (t + 1 < NK) load_tile((t + 1) * BK);
        compute(cur);
        if (t + 1 < NK) {
            store_tile(1 - cur);
            __syncthreads();
        }
    }

    #pragma unroll
    for (int j = 0; j < 2; ++j) {
        const int nloc = wc * 32 + j * 16 + l16;
        const float bias = bo[(size_t)e * HH + n0 + nloc];
        #pragma unroll
        for (int i = 0; i < 4; ++i) {
            #pragma unroll
            for (int v = 0; v < 8; ++v) {
                const int mloc = wr * 64 + i * 16 + g16 * 8 + v;
                out[((size_t)u * CC + (m0 + mloc)) * HH + (n0 + nloc)] =
                    acc[i][j][v] + bias;
            }
        }
    }
}

// ---------------------------------------------------------------------------
extern "C" void kernel_launch(void* const* d_in, const int* in_sizes, int n_in,
                              void* d_out, int out_size, void* d_ws, size_t ws_size,
                              hipStream_t stream) {
    const float* inp = (const float*)d_in[0];   // (G,E,C,H) f32
    const float* wi  = (const float*)d_in[1];   // (E,H,F)   f32
    const float* bi  = (const float*)d_in[2];   // (E,F)     f32
    const float* wo  = (const float*)d_in[3];   // (E,F,H)   f32
    const float* bo  = (const float*)d_in[4];   // (E,H)     f32
    float* out = (float*)d_out;                 // (G,E,C,H) f32
    __bf16* ws = (__bf16*)d_ws;

    const int totalU = 8 * EE;                             // G*E = 64 units
    const size_t perU = (size_t)CC * FF * sizeof(__bf16);  // 4 MB / unit
    int chunk = (ws_size >= perU) ? (int)(ws_size / perU) : 1;
    if (chunk > totalU) chunk = totalU;

    for (int u0 = 0; u0 < totalU; u0 += chunk) {
        const int cu = (totalU - u0 < chunk) ? (totalU - u0) : chunk;
        dim3 g1(FF / BN, CC / BM, cu);
        ffn_gemm1_gelu<<<g1, dim3(256), 0, stream>>>(inp, wi, bi, ws, u0);
        dim3 g2(HH / BN, CC / BM, cu);
        ffn_gemm2<<<g2, dim3(256), 0, stream>>>(ws, wo, bo, out, u0);
    }
}
